// GCNUnit_50998441672826
// MI455X (gfx1250) — matrix operand order repaired and checked by reading, not compiled, not run on hardware
//
#include <hip/hip_runtime.h>

// ---------------------------------------------------------------------------
// Fused ST-GCN unit for MI455X (gfx1250, wave32, WMMA).
//   conv1x1 + graph-aggregation folded into one [9600x1600]x[1600x1600] bf16
//   WMMA GEMM (f32 accum), then batch-norm stats + in-place normalize.
// ---------------------------------------------------------------------------

#define NB   32
#define CIN  64
#define COUT 64
#define KK   3
#define TT   300
#define VV   25

#define ROWS (NB * TT)            // 9600
#define COLS (COUT * VV)          // 1600 (output p = c*25+w)
#define KDIM (CIN * VV)           // 1600 (reduce  q = ci*25+v)
#define COLS_PAD 1664             // 13 * 128, padded rows of M
#define TOTAL_ELEMS (NB * COUT * TT * VV)  // 15,360,000

#define BM 128
#define BN 128
#define BK 32
#define LDSW 40                   // padded LDS row stride (elements): 80B, 16B aligned

typedef __attribute__((ext_vector_type(16))) __bf16 v16bf;
typedef __attribute__((ext_vector_type(8)))  float  v8f;

__device__ inline unsigned short f2bf(float x) {
    unsigned int u = __builtin_bit_cast(unsigned int, x);
    u += 0x7FFFu + ((u >> 16) & 1u);          // round-to-nearest-even
    return (unsigned short)(u >> 16);
}

// --- pack x [N,CIN,T,V] f32 -> Xb [n*T+t][ci*V+v] bf16 ----------------------
__global__ __launch_bounds__(256) void k_pack(const float* __restrict__ x,
                                              unsigned short* __restrict__ Xb) {
    int i = blockIdx.x * 256 + threadIdx.x;          // exactly TOTAL_ELEMS threads
    int n  = i / (CIN * TT * VV);
    int r  = i - n * (CIN * TT * VV);
    int ci = r / (TT * VV);
    int r2 = r - ci * (TT * VV);
    int t  = r2 / VV;
    int v  = r2 - t * VV;
    Xb[(size_t)(n * TT + t) * KDIM + ci * VV + v] = f2bf(x[i]);
}

// --- Mw[p][q] = sum_k conv_w[(k*64+c)*64+ci] * (A*imp)[k,v,w], bf16 ---------
__global__ __launch_bounds__(256) void k_build_m(const float* __restrict__ conv_w,
                                                 const float* __restrict__ A,
                                                 const float* __restrict__ imp,
                                                 unsigned short* __restrict__ Mw) {
    int i = blockIdx.x * 256 + threadIdx.x;
    if (i >= COLS_PAD * KDIM) return;
    int p = i / KDIM, q = i - p * KDIM;
    if (p >= COLS) { Mw[i] = 0; return; }            // zero padding rows
    int c = p / VV, w = p - c * VV;
    int ci = q / VV, v = q - ci * VV;
    float s = 0.f;
#pragma unroll
    for (int k = 0; k < KK; ++k) {
        int g = k * (VV * VV) + v * VV + w;
        s += conv_w[(k * COUT + c) * CIN + ci] * (A[g] * imp[g]);
    }
    Mw[i] = f2bf(s);
}

// --- bias[p=(c,w)] = sum_k conv_b[k*64+c] * sum_v (A*imp)[k,v,w] ------------
__global__ __launch_bounds__(256) void k_build_bias(const float* __restrict__ conv_b,
                                                    const float* __restrict__ A,
                                                    const float* __restrict__ imp,
                                                    float* __restrict__ bias) {
    int p = blockIdx.x * 256 + threadIdx.x;
    if (p >= COLS) return;
    int c = p / VV, w = p - c * VV;
    float s = 0.f;
#pragma unroll
    for (int k = 0; k < KK; ++k) {
        float gs = 0.f;
        for (int v = 0; v < VV; ++v) {
            int g = k * (VV * VV) + v * VV + w;
            gs += A[g] * imp[g];
        }
        s += conv_b[k * COUT + c] * gs;
    }
    bias[p] = s;
}

// --- GEMM: out(pre-BN, [N,C,T,V] layout) = Xb x Mw^T + bias -----------------
__global__ __launch_bounds__(256) void k_gemm(const unsigned short* __restrict__ Xb,
                                              const unsigned short* __restrict__ Mw,
                                              const float* __restrict__ bias,
                                              float* __restrict__ out) {
    __shared__ unsigned short As[BM * LDSW];   // [row r_local][k]
    __shared__ unsigned short Bs[BN * LDSW];   // [col p_local][k]

    const int tid = threadIdx.x;
    const int p0 = blockIdx.x * BN;            // 0..1536
    const int r0 = blockIdx.y * BM;            // 0..9472
    const int wv = tid >> 5, ln = tid & 31;
    const int lr = ln & 15, hi = ln >> 4;
    const int wrow = wv & 3, wcol = wv >> 2;   // 4x2 wave grid -> 32x64 per wave

    v8f acc[2][4];
#pragma unroll
    for (int mi = 0; mi < 2; ++mi)
#pragma unroll
        for (int ni = 0; ni < 4; ++ni)
            acc[mi][ni] = (v8f){0.f,0.f,0.f,0.f,0.f,0.f,0.f,0.f};

    // Each thread loads 16 bf16 (2x uint4) of A and of B per K-step.
    const int row  = tid >> 1;                 // 0..127
    const int seg  = (tid & 1) << 4;           // 0 or 16 (elements)
    const unsigned short* aptr = Xb + (size_t)(r0 + row) * KDIM + seg;
    const unsigned short* bptr = Mw + (size_t)(p0 + row) * KDIM + seg;  // rows<1664 ok
    unsigned short* asl = As + row * LDSW + seg;
    unsigned short* bsl = Bs + row * LDSW + seg;

    for (int kb = 0; kb < KDIM; kb += BK) {
        uint4 a0 = *(const uint4*)(aptr + kb);
        uint4 a1 = *(const uint4*)(aptr + kb + 8);
        uint4 b0 = *(const uint4*)(bptr + kb);
        uint4 b1 = *(const uint4*)(bptr + kb + 8);
        __syncthreads();
        *(uint4*)(asl)     = a0;  *(uint4*)(asl + 8) = a1;
        *(uint4*)(bsl)     = b0;  *(uint4*)(bsl + 8) = b1;
        __syncthreads();

        v16bf af[2], bfg[4];
        // A fragment: 16-bit A 16x32 layout — lanes 0-15: K 0..7,16..23; 16-31: +8
#pragma unroll
        for (int mi = 0; mi < 2; ++mi) {
            const unsigned int* pA =
                (const unsigned int*)(As + (wrow * 32 + mi * 16 + lr) * LDSW);
            union { v16bf v; unsigned int u[8]; } f;
#pragma unroll
            for (int j = 0; j < 4; ++j) {
                f.u[j]     = pA[hi * 4 + j];        // K = hi*8 + 2j
                f.u[4 + j] = pA[8 + hi * 4 + j];    // K = 16 + hi*8 + 2j
            }
            af[mi] = f.v;
        }
        // B fragment: 32x16 layout — same K order as A: lanes 0-15: K 0..7,16..23; 16-31: +8
#pragma unroll
        for (int ni = 0; ni < 4; ++ni) {
            const unsigned int* pB =
                (const unsigned int*)(Bs + (wcol * 64 + ni * 16 + lr) * LDSW);
            union { v16bf v; unsigned int u[8]; } f;
#pragma unroll
            for (int j = 0; j < 4; ++j) { f.u[j] = pB[hi * 4 + j]; f.u[4 + j] = pB[8 + hi * 4 + j]; }
            bfg[ni] = f.v;
        }
#pragma unroll
        for (int mi = 0; mi < 2; ++mi)
#pragma unroll
            for (int ni = 0; ni < 4; ++ni)
                acc[mi][ni] = __builtin_amdgcn_wmma_f32_16x16x32_bf16(
                    false, af[mi], false, bfg[ni], (short)0, acc[mi][ni],
                    false, false);
    }

    // Epilogue: +bias, scatter to [N,C,T,V]
#pragma unroll
    for (int ni = 0; ni < 4; ++ni) {
        int p = p0 + wcol * 64 + ni * 16 + lr;
        bool ok = (p < COLS);
        float bv = ok ? bias[p] : 0.f;
        int c = p / VV, w = p - c * VV;
#pragma unroll
        for (int mi = 0; mi < 2; ++mi) {
            int rbase = r0 + wrow * 32 + mi * 16 + hi * 8;   // D: lanes 0-15 M=j, 16-31 M=j+8
#pragma unroll
            for (int j = 0; j < 8; ++j) {
                if (ok) {
                    int r = rbase + j;
                    int n = r / TT, t = r - n * TT;
                    out[(size_t)((n * COUT + c) * TT + t) * VV + w] =
                        acc[mi][ni][j] + bv;
                }
            }
        }
    }
}

// --- per-channel batch stats (mean, rsqrt(var+eps)) -------------------------
__global__ __launch_bounds__(256) void k_stats(const float* __restrict__ out,
                                               float* __restrict__ stats) {
    __shared__ float ssum[256], ssq[256];
    const int c = blockIdx.x;
    float s = 0.f, q = 0.f;
    for (int i = threadIdx.x; i < NB * TT * VV; i += 256) {
        int n = i / (TT * VV), o = i - n * (TT * VV);
        float v = out[(size_t)(n * COUT + c) * (TT * VV) + o];
        s += v; q += v * v;
    }
    ssum[threadIdx.x] = s; ssq[threadIdx.x] = q;
    __syncthreads();
    for (int st = 128; st > 0; st >>= 1) {
        if (threadIdx.x < st) {
            ssum[threadIdx.x] += ssum[threadIdx.x + st];
            ssq[threadIdx.x]  += ssq[threadIdx.x + st];
        }
        __syncthreads();
    }
    if (threadIdx.x == 0) {
        const float inv = 1.f / (float)(NB * TT * VV);
        float mean = ssum[0] * inv;
        float var  = ssq[0] * inv - mean * mean;
        stats[c * 2]     = mean;
        stats[c * 2 + 1] = rsqrtf(var + 1e-5f);
    }
}

// --- in-place normalize + affine --------------------------------------------
__global__ __launch_bounds__(256) void k_norm(float* __restrict__ out,
                                              const float* __restrict__ stats,
                                              const float* __restrict__ gamma,
                                              const float* __restrict__ beta) {
    size_t i = (size_t)blockIdx.x * 256 + threadIdx.x;   // exactly TOTAL_ELEMS
    int c = (int)((i / (TT * VV)) & 63);
    float m = stats[c * 2], rs = stats[c * 2 + 1];
    out[i] = (out[i] - m) * rs * gamma[c] + beta[c];
}

extern "C" void kernel_launch(void* const* d_in, const int* in_sizes, int n_in,
                              void* d_out, int out_size, void* d_ws, size_t ws_size,
                              hipStream_t stream) {
    const float* x      = (const float*)d_in[0];
    const float* A      = (const float*)d_in[1];
    const float* conv_w = (const float*)d_in[2];
    const float* conv_b = (const float*)d_in[3];
    const float* imp    = (const float*)d_in[4];
    const float* gamma  = (const float*)d_in[5];
    const float* beta   = (const float*)d_in[6];
    float* out = (float*)d_out;

    char* ws = (char*)d_ws;
    const size_t XB_BYTES = (size_t)ROWS * KDIM * 2;            // 30,720,000
    const size_t MW_BYTES = (size_t)COLS_PAD * KDIM * 2;        //  5,324,800
    const size_t BI_BYTES = (size_t)COLS * 4;                   //      6,400
    unsigned short* Xb   = (unsigned short*)(ws);
    unsigned short* Mw   = (unsigned short*)(ws + XB_BYTES);
    float*          bias = (float*)(ws + XB_BYTES + MW_BYTES);
    float*          stat = (float*)(ws + XB_BYTES + MW_BYTES + BI_BYTES);
    (void)ws_size; (void)in_sizes; (void)n_in; (void)out_size;

    k_pack<<<TOTAL_ELEMS / 256, 256, 0, stream>>>(x, Xb);
    k_build_m<<<(COLS_PAD * KDIM + 255) / 256, 256, 0, stream>>>(conv_w, A, imp, Mw);
    k_build_bias<<<(COLS + 255) / 256, 256, 0, stream>>>(conv_b, A, imp, bias);
    k_gemm<<<dim3(COLS_PAD / BN, ROWS / BM), 256, 0, stream>>>(Xb, Mw, bias, out);
    k_stats<<<COUT, 256, 0, stream>>>(out, stat);
    k_norm<<<TOTAL_ELEMS / 256, 256, 0, stream>>>(out, stat, gamma, beta);
}
